// TemporalReasoningRNN_79302276153626
// MI455X (gfx1250) — matrix operand order repaired and checked by reading, not compiled, not run on hardware
//
#include <hip/hip_runtime.h>
#include <hip/hip_bf16.h>

// ---------------------------------------------------------------------------
// TemporalReasoningRNN for MI455X (gfx1250, wave32, WMMA bf16 16x16x32)
//
// Phases (all on `stream`, serialized by stream order):
//   1) convert W_out fp32 -> bf16 (8 MB, lives in L2; halves B-read traffic)
//   2) gather embeddings + x_gates GEMM (bf16 WMMA, fp32 accum, biases folded)
//   3) serial LSTM scan: 1 workgroup x 32 waves, W_hh^T in registers (bf16),
//      h tile in LDS, per-thread c-state in a register, 4 WMMAs/wave/step,
//      x_gates double-buffered in LDS via async global->LDS (ASYNCcnt) so the
//      next step's 16 KB fetch fully overlaps this step's compute
//   4) output projection: per-wave B slice resident in registers across 32
//      M-tiles (64 MB total B traffic), C staged through padded LDS, streamed
//      out with coalesced nontemporal b128 stores (524 MB => HBM store floor)
// ---------------------------------------------------------------------------

typedef __attribute__((ext_vector_type(16))) __bf16 v16bf;
typedef __attribute__((ext_vector_type(8)))  float  v8f;
typedef __attribute__((ext_vector_type(4)))  float  v4f;

#define BQ   8
#define SQ   512
#define DQ   128
#define HQ   128
#define G4   512           // 4*H
#define VOC  32000
#define ROWS (BQ * SQ)     // 4096
#define ASTRIDE 132        // LDS row stride (elems) for bank-conflict padding
#define CSTRIDE 132        // fp32 C-tile row stride (conflict-free store reads)

// Async global->LDS path (gfx1250 ASYNCcnt engine), guarded so the build stays
// green if the toolchain lacks the builtins (falls back to early VMEM loads).
// Probe-learned prototype:
//   void __builtin_amdgcn_global_load_async_to_lds_b128(
//       v4i* gsrc, __attribute__((address_space(3))) v4i* ldsdst,
//       imm int offset, imm int cpol)
#if defined(__gfx1250__) && \
    __has_builtin(__builtin_amdgcn_global_load_async_to_lds_b128) && \
    __has_builtin(__builtin_amdgcn_s_wait_asynccnt)
#define USE_ASYNC_LDS 1
typedef int v4i_ldst __attribute__((vector_size(16)));
typedef __attribute__((address_space(3))) v4i_ldst* lds_v4i_t;
#else
#define USE_ASYNC_LDS 0
#endif

// ---- WMMA operand loaders --------------------------------------------------
// A (16x32 bf16, MxK) from an LDS tile with row stride ASTRIDE.
// Contiguous K order used here: lanes 0-15 hold K 0..15 pairs, lanes 16-31 hold K 16..31 pairs:
// v[0..15] = A[m][kc*32 + hi*16 + 0..15] (matches B's contiguous K order)
__device__ __forceinline__ v16bf load_A_lds(const __bf16* A, int kc, int lane) {
  int m  = lane & 15;
  int hi = lane >> 4;
  const __bf16* p0 = A + m * ASTRIDE + kc * 32 + hi * 16;
  union { unsigned long long u[4]; v16bf v; } r;
  r.u[0] = *(const unsigned long long*)(p0);
  r.u[1] = *(const unsigned long long*)(p0 + 4);
  r.u[2] = *(const unsigned long long*)(p0 + 8);
  r.u[3] = *(const unsigned long long*)(p0 + 12);
  return r.v;
}

// B (32x16 bf16, KxN) straight from global bf16 weights stored [N][K] row-major
// (i.e. W[n][k]); column n of B == row n of W is K-contiguous, so each lane
// does ONE 32-byte contiguous load: lanes 0-15 -> K 0..15, lanes 16-31 -> 16..31.
__device__ __forceinline__ v16bf load_B_gbl(const __bf16* Wn_base, int kc, int lane) {
  int n  = lane & 15;
  int kg = lane >> 4;
  return *(const v16bf*)(Wn_base + n * DQ + kc * 32 + kg * 16);
}

// Same but converting from fp32 weights on the fly (cheap, done once).
__device__ __forceinline__ v16bf load_B_gbl_f32(const float* Wn_base, int kc, int lane) {
  int n  = lane & 15;
  int kg = lane >> 4;
  const float* p = Wn_base + n * DQ + kc * 32 + kg * 16;
  v16bf b;
#pragma unroll
  for (int i = 0; i < 16; ++i) b[i] = (__bf16)p[i];
  return b;
}

// ---- 1) W_out fp32 -> bf16 -------------------------------------------------
__global__ void k_convert_wout(const float* __restrict__ w, __bf16* __restrict__ o) {
  int base = (blockIdx.x * 256 + threadIdx.x) * 8;   // 4,096,000 elems total
  float4 a = *(const float4*)(w + base);
  float4 b = *(const float4*)(w + base + 4);
  __bf16 t[8] = {(__bf16)a.x,(__bf16)a.y,(__bf16)a.z,(__bf16)a.w,
                 (__bf16)b.x,(__bf16)b.y,(__bf16)b.z,(__bf16)b.w};
  *(ulonglong2*)(o + base) = *(ulonglong2*)t;
}

// ---- 2) embeddings gather + x_gates = emb @ W_ih^T + (b_ih + b_hh) ---------
// x_gates layout: [S][B][4H] fp32 (row = s*8+b), matching the scan order.
__global__ void k_xgates(const int*   __restrict__ seq,   // [B,S]
                         const float* __restrict__ emb,   // [VOC,D]
                         const float* __restrict__ Wih,   // [4H,D]
                         const float* __restrict__ bih,
                         const float* __restrict__ bhh,
                         float* __restrict__ xg) {        // [S*B, 4H]
  __shared__ __bf16 atile[16 * ASTRIDE];
  const int mt   = blockIdx.x;          // 256 row-tiles of 16
  const int ngrp = blockIdx.y;          // 8 groups of 64 cols
  const int t    = threadIdx.x;         // 128 threads = 4 waves
  const int lane = t & 31, wave = t >> 5;

  // gather 16 rows x 128 dims of embeddings, convert to bf16
  for (int i = t; i < 16 * DQ; i += 128) {
    int m = i >> 7, d = i & 127;
    int row = mt * 16 + m;              // row = s*8 + b
    int s = row >> 3, b = row & 7;
    int tok = seq[b * SQ + s];
    atile[m * ASTRIDE + d] = (__bf16)emb[(long long)tok * DQ + d];
  }
  __syncthreads();

  const int nbase = (ngrp * 4 + wave) * 16;
  v8f acc = {};
#pragma unroll
  for (int kc = 0; kc < 4; ++kc) {
    v16bf a = load_A_lds(atile, kc, lane);
    v16bf b = load_B_gbl_f32(Wih + nbase * DQ, kc, lane);
    acc = __builtin_amdgcn_wmma_f32_16x16x32_bf16(false, a, false, b,
                                                  (short)0, acc, false, false);
  }
  // C layout: lane l holds N = l&15; VGPR r holds M = r + 8*(l>>4)
  const int n  = lane & 15, hi = lane >> 4;
  const int ng = nbase + n;
  const float bias = bih[ng] + bhh[ng];
#pragma unroll
  for (int r = 0; r < 8; ++r) {
    int row = mt * 16 + r + hi * 8;
    xg[row * G4 + ng] = acc[r] + bias;
  }
}

// ---- 3) serial LSTM scan ---------------------------------------------------
// One workgroup, 32 waves. Wave w owns gate columns [w*16, w*16+16) of W_hh^T,
// held as 4 v16bf register chunks for all 512 steps. h (bf16, 16 rows padded)
// and z (fp32 [8][512]) live in LDS. Thread t owns cell (b = t/128, j = t%128)
// and keeps c in a register across the whole scan. x_gates is double-buffered
// in LDS: each thread issues one async b128 global->LDS transfer for step s+1
// right after the mid-step barrier; completion = s_wait_asynccnt 0 + barrier.
__global__ void k_lstm(const float* __restrict__ xg,     // [S*B, 4H]
                       const float* __restrict__ Whh,    // [4H, H]
                       __bf16* __restrict__ hsb,         // [B*S, H] bf16 (row = b*512+s)
                       float*  __restrict__ out_tail) {  // h_n then c_n (fp32)
  __shared__ __bf16 hbuf[16 * ASTRIDE];
  __shared__ float  zbuf[BQ * G4];
#if USE_ASYNC_LDS
  __shared__ float  xbuf[2][BQ * G4];   // 2 x 16 KB double buffer
#endif
  const int t = threadIdx.x;            // 1024 threads = 32 waves
  const int lane = t & 31, wave = t >> 5;

  for (int i = t; i < 16 * ASTRIDE; i += 1024) hbuf[i] = (__bf16)0.0f;

  // persistent B operand: W_hh^T slice, bf16, in registers
  v16bf bch[4];
#pragma unroll
  for (int kc = 0; kc < 4; ++kc)
    bch[kc] = load_B_gbl_f32(Whh + (wave * 16) * HQ, kc, lane);

  const int b_idx = t >> 7;             // 0..7
  const int j_idx = t & 127;            // 0..127
  const int n_l   = lane & 15;
  const int hi_l  = lane >> 4;
  const int ngc   = wave * 16 + n_l;    // gate column this lane publishes
  float c_reg = 0.0f;

#if USE_ASYNC_LDS
  // kick off step 0's x_gates slice (16 KB = 1024 threads x one b128 each)
  __builtin_amdgcn_global_load_async_to_lds_b128(
      (v4i_ldst*)(xg + t * 4), (lds_v4i_t)&xbuf[0][t * 4], 0, 0);
  __builtin_amdgcn_s_wait_asynccnt(0);
#endif
  __syncthreads();

  for (int s = 0; s < SQ; ++s) {
    // x_gates for this step: cheap LDS reads (async path) or early VMEM loads
    float xv[BQ];
#if USE_ASYNC_LDS
    const int cur = s & 1;
#pragma unroll
    for (int r = 0; r < BQ; ++r) xv[r] = xbuf[cur][r * G4 + ngc];
#else
#pragma unroll
    for (int r = 0; r < BQ; ++r) xv[r] = xg[(s * BQ + r) * G4 + ngc];
#endif

    v8f acc = {};
#pragma unroll
    for (int kc = 0; kc < 4; ++kc) {
      v16bf a = load_A_lds(hbuf, kc, lane);
      acc = __builtin_amdgcn_wmma_f32_16x16x32_bf16(false, a, false, bch[kc],
                                                    (short)0, acc, false, false);
    }
    // publish z for this wave's 16 gate columns (valid rows b=0..7 live in
    // VGPRs 0..7 of lanes 0..15; lanes 16..31 hold the zero-padded rows)
    if (hi_l == 0) {
#pragma unroll
      for (int r = 0; r < BQ; ++r)
        zbuf[r * G4 + ngc] = acc[r] + xv[r];
    }
    __syncthreads();

    // overlap the next step's 16 KB x_gates fetch with this step's tail
#if USE_ASYNC_LDS
    if (s + 1 < SQ)
      __builtin_amdgcn_global_load_async_to_lds_b128(
          (v4i_ldst*)(xg + (s + 1) * (BQ * G4) + t * 4),
          (lds_v4i_t)&xbuf[cur ^ 1][t * 4], 0, 0);
#else
    if (s + 1 < SQ) __builtin_prefetch(&xg[(s + 1) * (BQ * G4) + t * 4], 0, 1);
#endif

    // gate nonlinearities; PyTorch packing i,f,g,o
    const float zi = zbuf[b_idx * G4 + j_idx];
    const float zf = zbuf[b_idx * G4 + HQ + j_idx];
    const float zg = zbuf[b_idx * G4 + 2 * HQ + j_idx];
    const float zo = zbuf[b_idx * G4 + 3 * HQ + j_idx];
    const float ig = 1.0f / (1.0f + __expf(-zi));
    const float fg = 1.0f / (1.0f + __expf(-zf));
    const float gg = tanhf(zg);
    const float og = 1.0f / (1.0f + __expf(-zo));
    c_reg = fg * c_reg + ig * gg;
    const float h = og * tanhf(c_reg);

    const __bf16 hb = (__bf16)h;
    hbuf[b_idx * ASTRIDE + j_idx] = hb;                    // next step's A
    hsb[(b_idx * SQ + s) * HQ + j_idx] = hb;               // for projection
    if (s == SQ - 1) {
      out_tail[b_idx * HQ + j_idx]           = h;          // h_n
      out_tail[BQ * HQ + b_idx * HQ + j_idx] = c_reg;      // c_n
    }
#if USE_ASYNC_LDS
    __builtin_amdgcn_s_wait_asynccnt(0);   // own transfers done; barrier makes
#endif                                     // it workgroup-wide
    __syncthreads();
  }
}

// ---- 4) predictions = hs @ W_out^T + b_out ---------------------------------
// Store-bandwidth bound (524 MB fp32 out). Each block owns a 128-col slice:
// B operand lives in registers across 32 M-tiles (total B traffic 64 MB),
// C is staged through padded LDS and streamed out with coalesced NT stores.
__global__ void k_proj(const __bf16* __restrict__ hsb,    // [4096,128]
                       const __bf16* __restrict__ woutb,  // [32000,128]
                       const float*  __restrict__ bout,   // [32000]
                       float* __restrict__ pred) {        // [4096,32000]
  __shared__ __bf16 atile[16 * ASTRIDE];
  __shared__ float  ctile[16 * CSTRIDE];
  const int ngrp   = blockIdx.x;        // 250 col-groups of 128
  const int mchunk = blockIdx.y;        // 8 row-chunks of 32 tiles (512 rows)
  const int t      = threadIdx.x;       // 256 threads = 8 waves
  const int lane = t & 31, wave = t >> 5;

  // per-wave resident B slice (4 x v16bf = 32 VGPRs) + bias
  const int nbase = ngrp * 128 + wave * 16;
  v16bf bch[4];
#pragma unroll
  for (int kc = 0; kc < 4; ++kc)
    bch[kc] = load_B_gbl(woutb + (long long)nbase * DQ, kc, lane);
  const float bias = bout[nbase + (lane & 15)];

  // store-phase mapping: thread t handles 8 consecutive floats of the C tile
  const int srow = (t * 8) >> 7;        // 0..15
  const int scol = (t * 8) & 127;       // 0,8,...,120

  for (int i = 0; i < 32; ++i) {
    const int mt = mchunk * 32 + i;
    if (i + 1 < 32) __builtin_prefetch(hsb + (mt + 1) * (16 * HQ), 0, 1);

    // stage A tile (16x128 bf16 = 4 KB)
    for (int j = t; j < 16 * HQ; j += 256) {
      int m = j >> 7, d = j & 127;
      atile[m * ASTRIDE + d] = hsb[mt * (16 * HQ) + j];
    }
    __syncthreads();

    v8f acc = {};
#pragma unroll
    for (int kc = 0; kc < 4; ++kc) {
      v16bf a = load_A_lds(atile, kc, lane);
      acc = __builtin_amdgcn_wmma_f32_16x16x32_bf16(false, a, false, bch[kc],
                                                    (short)0, acc, false, false);
    }
    // scatter C into padded LDS (halves land 32 banks apart)
    {
      const int n = lane & 15, hi = lane >> 4;
#pragma unroll
      for (int r = 0; r < 8; ++r)
        ctile[(r + hi * 8) * CSTRIDE + wave * 16 + n] = acc[r] + bias;
    }
    __syncthreads();

    // coalesced nontemporal stores: 512B-contiguous row segments per wave
    {
      const float* src = ctile + srow * CSTRIDE + scol;
      long long rg = (long long)(mt * 16 + srow);
      float* dst = pred + rg * VOC + ngrp * 128 + scol;
      v4f v0 = *(const v4f*)(src);
      v4f v1 = *(const v4f*)(src + 4);
      __builtin_nontemporal_store(v0, (v4f*)dst);
      __builtin_nontemporal_store(v1, (v4f*)(dst + 4));
    }
    // next iteration's barrier-after-staging protects ctile reuse; atile is
    // only rewritten after this point, all its reads finished pre-ctile-write
  }
}

// ---------------------------------------------------------------------------
extern "C" void kernel_launch(void* const* d_in, const int* in_sizes, int n_in,
                              void* d_out, int out_size, void* d_ws, size_t ws_size,
                              hipStream_t stream) {
  const int*   seq  = (const int*)  d_in[0];
  const float* emb  = (const float*)d_in[1];
  const float* Wih  = (const float*)d_in[2];
  const float* Whh  = (const float*)d_in[3];
  const float* bih  = (const float*)d_in[4];
  const float* bhh  = (const float*)d_in[5];
  const float* Wout = (const float*)d_in[6];
  const float* bout = (const float*)d_in[7];
  float* out = (float*)d_out;

  char* ws = (char*)d_ws;
  __bf16* woutb = (__bf16*)(ws);                               //  8,192,000 B
  float*  xg    = (float*) (ws + 8192000);                     //  8,388,608 B
  __bf16* hsb   = (__bf16*)(ws + 8192000 + 8388608);           //  1,048,576 B

  const long long PRED = (long long)ROWS * VOC;                // 131,072,000

  k_convert_wout<<<dim3(VOC * DQ / (256 * 8)), 256, 0, stream>>>(Wout, woutb);
  k_xgates      <<<dim3(ROWS / 16, G4 / 64), 128, 0, stream>>>(seq, emb, Wih, bih, bhh, xg);
  k_lstm        <<<dim3(1), 1024, 0, stream>>>(xg, Whh, hsb, out + PRED);
  k_proj        <<<dim3(VOC / 128, ROWS / 512), 256, 0, stream>>>(hsb, woutb, bout, out);
}
